// TCWindowAttention_81990925681416
// MI455X (gfx1250) — compile-verified, matrix operand order checked by reading, not run on hardware
//
#include <hip/hip_runtime.h>
#include <hip/hip_bf16.h>
#include <math.h>

// ---------------- problem constants (fixed by setup_inputs) ----------------
#define B_      4
#define N_      784
#define C_      384
#define N0_     3136
#define H_      56
#define W_      56
#define HW_     3136          // H_*W_
#define NHEADS_ 8
#define DH_     48            // C_/NHEADS_
#define NWIN_   64            // 8x8 windows of 7x7
#define KWIN_   49
#define MPAD2_  3152          // (HW_+1) padded up to multiple of 16 (197*16)

typedef _Float16 half_t;
typedef __attribute__((ext_vector_type(16))) _Float16 v16h;
typedef __attribute__((ext_vector_type(8)))  _Float16 v8h;
typedef __attribute__((ext_vector_type(8)))  float    v8f;

// round-half-to-even grid snap (matches jnp.round)
__device__ __forceinline__ void grid_idx(float lx, float ly, int& xg, int& yg) {
    float fx = 0.5f * (lx + 1.0f) * (float)W_ - 0.5f;
    float fy = 0.5f * (ly + 1.0f) * (float)H_ - 0.5f;
    int x = __float2int_rn(fx);
    int y = __float2int_rn(fy);
    xg = min(max(x, 0), W_ - 1);
    yg = min(max(y, 0), H_ - 1);
}

// ---------------- conversion / transpose helpers ----------------
__global__ void convert_f32_to_f16(const float* __restrict__ in,
                                   half_t* __restrict__ out, int n) {
    for (int i = blockIdx.x * blockDim.x + threadIdx.x; i < n;
         i += gridDim.x * blockDim.x)
        out[i] = (half_t)in[i];
}

// wT[n*K + k] = (f16) w[k*Nout + n]   (row-major KxN  ->  row-major NxK)
__global__ void transpose_to_f16(const float* __restrict__ w,
                                 half_t* __restrict__ wT, int K, int Nout) {
    int total = K * Nout;
    for (int i = blockIdx.x * blockDim.x + threadIdx.x; i < total;
         i += gridDim.x * blockDim.x) {
        int n = i / K;
        int k = i - n * K;
        wT[i] = (half_t)w[(size_t)k * Nout + n];
    }
}

// ---------------- scatter: smap accumulation (token2map) ----------------
// one block per (b, i) source token; 385 channels scatter-added with atomics
__global__ void scatter_smap_kernel(const float* __restrict__ x_source,
                                    const float* __restrict__ conf_source,
                                    const float* __restrict__ loc_orig,
                                    const int*   __restrict__ idx_agg_source,
                                    float* __restrict__ acc,   // [B, HW, 385]
                                    float* __restrict__ cnt) { // [B, HW]
    int bi = blockIdx.x;            // b*N0 + i
    int b  = bi / N0_;
    float lx = loc_orig[(size_t)bi * 2 + 0];
    float ly = loc_orig[(size_t)bi * 2 + 1];
    int xg, yg; grid_idx(lx, ly, xg, yg);
    int pix = yg * W_ + xg;
    int j = idx_agg_source[bi];
    const float* srcrow = x_source + ((size_t)b * N_ + j) * C_;
    float* accrow = acc + ((size_t)b * HW_ + pix) * 385;
    for (int c = threadIdx.x; c < C_; c += blockDim.x)
        atomicAdd(&accrow[c], srcrow[c]);
    if (threadIdx.x == 0) {
        atomicAdd(&accrow[C_], conf_source[(size_t)b * N_ + j]);
        atomicAdd(&cnt[(size_t)b * HW_ + pix], 1.0f);
    }
}

// ---------------- scatter: per-token window vote ----------------
__global__ void scatter_pw_kernel(const float* __restrict__ loc_orig,
                                  const int*   __restrict__ idx_agg,
                                  const float* __restrict__ agg_weight,
                                  float* __restrict__ pw) { // [B, N, 64]
    int tid = blockIdx.x * blockDim.x + threadIdx.x;
    if (tid >= B_ * N0_) return;
    int b = tid / N0_;
    float lx = loc_orig[(size_t)tid * 2 + 0];
    float ly = loc_orig[(size_t)tid * 2 + 1];
    int xg, yg; grid_idx(lx, ly, xg, yg);
    int idxt = (yg / 7) * 8 + (xg / 7);     // window_map[yg, xg]
    int n = idx_agg[tid];
    atomicAdd(&pw[((size_t)b * N_ + n) * NWIN_ + idxt], agg_weight[tid]);
}

__global__ void argmax_window_kernel(const float* __restrict__ pw,
                                     int* __restrict__ idx_window) {
    int tid = blockIdx.x * blockDim.x + threadIdx.x;
    if (tid >= B_ * N_) return;
    const float* row = pw + (size_t)tid * NWIN_;
    float best = row[0]; int bi = 0;
    for (int w = 1; w < NWIN_; ++w) {
        float v = row[w];
        if (v > best) { best = v; bi = w; }   // strict > keeps first max (jnp.argmax)
    }
    idx_window[tid] = bi;
}

// ---------------- finalize smap -> xs (f16) and cs ----------------
__global__ void build_xs_kernel(const float* __restrict__ acc,
                                const float* __restrict__ cnt,
                                half_t* __restrict__ xs_h,  // [B*MPAD2, C]
                                float*  __restrict__ cs) {  // [B*MPAD2]
    int row = blockIdx.x;           // b*MPAD2 + p
    int b = row / MPAD2_;
    int p = row - b * MPAD2_;
    half_t* xrow = xs_h + (size_t)row * C_;
    if (p < HW_) {
        float inv = 1.0f / (cnt[(size_t)b * HW_ + p] + 1e-6f);
        const float* arow = acc + ((size_t)b * HW_ + p) * 385;
        for (int c = threadIdx.x; c < C_; c += blockDim.x)
            xrow[c] = (half_t)(arow[c] * inv);
        if (threadIdx.x == 0) cs[row] = arow[C_] * inv;
    } else {
        for (int c = threadIdx.x; c < C_; c += blockDim.x)
            xrow[c] = (half_t)0.0f;
        if (threadIdx.x == 0) cs[row] = -INFINITY;  // appended row (+ pad rows)
    }
}

// ---------------- WMMA GEMM: C = A(f16, MxK) @ BT^T(f16, NoutxK) + bias ----
// one wave per 16x16 output tile; K-loop in steps of 32 (v_wmma_f32_16x16x32_f16)
__global__ __launch_bounds__(256) void wmma_gemm_f16(
    const half_t* __restrict__ A,     // row-major M x K
    const half_t* __restrict__ BT,    // row-major Nout x K (i.e. B column-major)
    const float*  __restrict__ bias,  // [Nout]
    float* __restrict__ Cout,         // row-major M x Nout
    int M, int K, int Nout) {
    int wave = threadIdx.x >> 5;
    int lane = threadIdx.x & 31;
    int tilesN = Nout >> 4;
    int totalTiles = (M >> 4) * tilesN;
    int t = blockIdx.x * 8 + wave;
    if (t >= totalTiles) return;
    int mt = t / tilesN;
    int nt = t - mt * tilesN;

    // 16-bit A/B fragment layout (ISA 7.12.2): lanes 0-15 k-base 0, 16-31 k-base 8;
    // halves [0..7] = K[kb..kb+7], halves [8..15] = K[kb+16..kb+23]
    int l15 = lane & 15;
    int kb  = (lane >> 4) << 3;
    const half_t* Ap = A  + (size_t)((mt << 4) + l15) * K + kb;
    const half_t* Bp = BT + (size_t)((nt << 4) + l15) * K + kb;

    v8f c = {};
    for (int k0 = 0; k0 < K; k0 += 32) {
        v8h alo = *(const v8h*)(Ap + k0);
        v8h ahi = *(const v8h*)(Ap + k0 + 16);
        v8h blo = *(const v8h*)(Bp + k0);
        v8h bhi = *(const v8h*)(Bp + k0 + 16);
        v16h a, bb;
        #pragma unroll
        for (int i = 0; i < 8; ++i) {
            a[i] = alo[i];  a[i + 8] = ahi[i];
            bb[i] = blo[i]; bb[i + 8] = bhi[i];
        }
        c = __builtin_amdgcn_wmma_f32_16x16x32_f16(
                false, a, false, bb, (short)0, c, false, false);
    }

    // C/D layout: n = nbase + (lane&15); VGPR r -> M = mbase + r (+8 for lanes 16-31)
    int n    = (nt << 4) + l15;
    int mrow = (mt << 4) + ((lane >> 4) << 3);
    float bv = bias[n];
    #pragma unroll
    for (int r = 0; r < 8; ++r)
        Cout[(size_t)(mrow + r) * Nout + n] = c[r] + bv;
}

// ---------------- windowed attention ----------------
// one block per (b, n) query token; wave w handles head w (wave32, 8 waves)
__global__ __launch_bounds__(256) void window_attn_kernel(
    const float* __restrict__ q,        // [B*N, C]
    const float* __restrict__ kv,       // [B*MPAD2, 2C]  (k: [h*48..], v: [384+h*48..])
    const float* __restrict__ cs,       // [B*MPAD2]
    const int*   __restrict__ idx_window, // [B*N]
    half_t* __restrict__ out_h) {       // [B*N, C]
    __shared__ float sq[NHEADS_][DH_];
    __shared__ float sw[NHEADS_][NWIN_];
    __shared__ float sconf[KWIN_ + 15];
    __shared__ int   spix[KWIN_ + 15];

    const float scale = 0.14433756729740643f;  // 48^-0.5
    int bn = blockIdx.x;
    int b  = bn / N_;
    int wave = threadIdx.x >> 5;
    int lane = threadIdx.x & 31;
    int h = wave;

    for (int dd = threadIdx.x; dd < C_; dd += blockDim.x)
        sq[dd / DH_][dd % DH_] = q[(size_t)bn * C_ + dd] * scale;

    if (threadIdx.x < KWIN_) {
        int wIdx = idx_window[bn];
        int wy = wIdx >> 3, wx = wIdx & 7;
        int r  = threadIdx.x / 7, cc = threadIdx.x - 7 * (threadIdx.x / 7);
        int p  = (wy * 7 + r) * W_ + (wx * 7 + cc);   // token_window (no padding)
        spix[threadIdx.x]  = p;
        sconf[threadIdx.x] = cs[(size_t)b * MPAD2_ + p];
    }
    __syncthreads();

    // logits: lanes over keys (kk = lane, lane+32)
    float lg0 = -3.0e38f, lg1 = -3.0e38f;
    {
        int kk = lane;
        int p = spix[kk];
        const float* krow = kv + ((size_t)b * MPAD2_ + p) * (2 * C_) + h * DH_;
        float s = 0.f;
        #pragma unroll
        for (int dd = 0; dd < DH_; ++dd) s += sq[h][dd] * krow[dd];
        lg0 = s + sconf[kk];
    }
    if (lane + 32 < KWIN_) {
        int kk = lane + 32;
        int p = spix[kk];
        const float* krow = kv + ((size_t)b * MPAD2_ + p) * (2 * C_) + h * DH_;
        float s = 0.f;
        #pragma unroll
        for (int dd = 0; dd < DH_; ++dd) s += sq[h][dd] * krow[dd];
        lg1 = s + sconf[kk];
    }

    // wave32 softmax reductions
    float m = fmaxf(lg0, lg1);
    for (int off = 16; off > 0; off >>= 1)
        m = fmaxf(m, __shfl_xor(m, off, 32));
    float e0 = __expf(lg0 - m);
    float e1 = (lane + 32 < KWIN_) ? __expf(lg1 - m) : 0.f;
    float ssum = e0 + e1;
    for (int off = 16; off > 0; off >>= 1)
        ssum += __shfl_xor(ssum, off, 32);
    float inv = 1.0f / ssum;
    sw[h][lane] = e0 * inv;
    if (lane + 32 < KWIN_) sw[h][lane + 32] = e1 * inv;
    __syncthreads();

    // out = sum_kk w_kk * v[kk][:] ; lanes over dd (dd = lane, lane+32 for lane<16)
    float acc0 = 0.f, acc1 = 0.f;
    for (int kk = 0; kk < KWIN_; ++kk) {
        int p = spix[kk];
        const float* vrow = kv + ((size_t)b * MPAD2_ + p) * (2 * C_) + C_ + h * DH_;
        float wgt = sw[h][kk];
        acc0 += wgt * vrow[lane];
        if (lane < 16) acc1 += wgt * vrow[32 + lane];
    }
    out_h[(size_t)bn * C_ + h * DH_ + lane] = (half_t)acc0;
    if (lane < 16)
        out_h[(size_t)bn * C_ + h * DH_ + 32 + lane] = (half_t)acc1;
}

// ---------------- host launch ----------------
extern "C" void kernel_launch(void* const* d_in, const int* in_sizes, int n_in,
                              void* d_out, int out_size, void* d_ws, size_t ws_size,
                              hipStream_t stream) {
    (void)in_sizes; (void)n_in; (void)out_size; (void)ws_size;

    const float* x           = (const float*)d_in[0];
    const float* loc_orig    = (const float*)d_in[1];
    const int*   idx_agg     = (const int*)  d_in[2];
    const float* agg_weight  = (const float*)d_in[3];
    const float* x_source    = (const float*)d_in[4];
    const int*   idx_agg_src = (const int*)  d_in[5];
    const float* conf_source = (const float*)d_in[6];
    const float* q_w    = (const float*)d_in[9];
    const float* q_b    = (const float*)d_in[10];
    const float* kv_w   = (const float*)d_in[11];
    const float* kv_b   = (const float*)d_in[12];
    const float* proj_w = (const float*)d_in[13];
    const float* proj_b = (const float*)d_in[14];
    float* out = (float*)d_out;

    // carve workspace (256B-aligned slots)
    char* base = (char*)d_ws;
    size_t off = 0;
    auto carve = [&](size_t bytes) {
        size_t cur = off;
        off += (bytes + 255) & ~(size_t)255;
        return (void*)(base + cur);
    };
    float*  qbuf   = (float*) carve((size_t)B_ * N_ * C_ * 4);          // q = x@q_w+q_b
    float*  acc    = (float*) carve((size_t)B_ * HW_ * 385 * 4);        // smap accum
    float*  cnt    = (float*) carve((size_t)B_ * HW_ * 4);
    float*  cs     = (float*) carve((size_t)B_ * MPAD2_ * 4);           // conf map
    float*  kvbuf  = (float*) carve((size_t)B_ * MPAD2_ * 2 * C_ * 4);  // k|v
    float*  pw     = (float*) carve((size_t)B_ * N_ * NWIN_ * 4);
    half_t* x_h    = (half_t*)carve((size_t)B_ * N_ * C_ * 2);
    half_t* xs_h   = (half_t*)carve((size_t)B_ * MPAD2_ * C_ * 2);
    half_t* attn_h = (half_t*)carve((size_t)B_ * N_ * C_ * 2);
    half_t* qwT    = (half_t*)carve((size_t)C_ * C_ * 2);
    half_t* kvwT   = (half_t*)carve((size_t)C_ * 2 * C_ * 2);
    half_t* pwT    = (half_t*)carve((size_t)C_ * C_ * 2);
    int*    idxwin = (int*)   carve((size_t)B_ * N_ * 4);

    // zero the scatter accumulators (graph-capturable async memsets)
    hipMemsetAsync(acc, 0, (size_t)B_ * HW_ * 385 * 4, stream);
    hipMemsetAsync(cnt, 0, (size_t)B_ * HW_ * 4, stream);
    hipMemsetAsync(pw,  0, (size_t)B_ * N_ * NWIN_ * 4, stream);

    // conversions / weight transposes
    convert_f32_to_f16<<<1024, 256, 0, stream>>>(x, x_h, B_ * N_ * C_);
    transpose_to_f16<<<(C_ * C_ + 255) / 256, 256, 0, stream>>>(q_w, qwT, C_, C_);
    transpose_to_f16<<<(C_ * 2 * C_ + 255) / 256, 256, 0, stream>>>(kv_w, kvwT, C_, 2 * C_);
    transpose_to_f16<<<(C_ * C_ + 255) / 256, 256, 0, stream>>>(proj_w, pwT, C_, C_);

    // scatter phase
    scatter_smap_kernel<<<B_ * N0_, 128, 0, stream>>>(x_source, conf_source,
                                                      loc_orig, idx_agg_src,
                                                      acc, cnt);
    scatter_pw_kernel<<<(B_ * N0_ + 255) / 256, 256, 0, stream>>>(loc_orig, idx_agg,
                                                                  agg_weight, pw);
    argmax_window_kernel<<<(B_ * N_ + 255) / 256, 256, 0, stream>>>(pw, idxwin);

    // q GEMM: (3136 x 384) @ (384 x 384)
    {
        int tiles = ((B_ * N_) >> 4) * (C_ >> 4);
        wmma_gemm_f16<<<(tiles + 7) / 8, 256, 0, stream>>>(x_h, qwT, q_b, qbuf,
                                                           B_ * N_, C_, C_);
    }

    // smap -> xs (f16) + cs
    build_xs_kernel<<<B_ * MPAD2_, 128, 0, stream>>>(acc, cnt, xs_h, cs);

    // kv GEMM: (12608 x 384) @ (384 x 768)
    {
        int tiles = ((B_ * MPAD2_) >> 4) * ((2 * C_) >> 4);
        wmma_gemm_f16<<<(tiles + 7) / 8, 256, 0, stream>>>(xs_h, kvwT, kv_b, kvbuf,
                                                           B_ * MPAD2_, C_, 2 * C_);
    }

    // windowed attention
    window_attn_kernel<<<B_ * N_, 256, 0, stream>>>(qbuf, kvbuf, cs, idxwin, attn_h);

    // output projection: (3136 x 384) @ (384 x 384) -> d_out (fp32)
    {
        int tiles = ((B_ * N_) >> 4) * (C_ >> 4);
        wmma_gemm_f16<<<(tiles + 7) / 8, 256, 0, stream>>>(attn_h, pwT, proj_b, out,
                                                           B_ * N_, C_, C_);
    }
}
